// MultiscaleDecomposedConv1D_3753801417263
// MI455X (gfx1250) — matrix operand order repaired and checked by reading, not compiled, not run on hardware
//
#include <hip/hip_runtime.h>

typedef __attribute__((ext_vector_type(16))) __bf16 v16bf;
typedef __attribute__((ext_vector_type(8)))  float  v8f;

constexpr int S_ = 8, T_ = 4096, CIN = 128, COUT = 256;
constexpr int KSIZE_T = 9;
constexpr int KTOT = KSIZE_T * CIN;   // 1152
constexpr int NKT  = KTOT / 32;       // 36 K-tiles of 32
constexpr int NNT  = COUT / 16;       // 16 N-tiles
constexpr int MT   = 128;             // time tile per workgroup
constexpr int LDSROW = 136;           // padded LDS row stride (bf16 elems) -> 272B, conflict-free
constexpr float PI_F = 3.14159265358979323846f;

__device__ __constant__ float d_scales[8] = {-1.5f,-1.0f,-0.5f,0.0f,0.5f,1.0f,1.5f,2.0f};

__device__ inline unsigned short f2bf(float f) {
  unsigned int u = __float_as_uint(f);
  u += 0x7FFF + ((u >> 16) & 1);      // round-to-nearest-even
  return (unsigned short)(u >> 16);
}

// ---------------------------------------------------------------------------
// Weight synthesis: Fourier basis -> bf16 weights, pre-swizzled into the WMMA
// B-fragment lane layout. Fragment (s, kt, nt) = 32 lanes x 16 bf16 (1KB).
// 32x16 bf16 B, K order used here (same as A): lane<16 -> N=lane, K = e (0..15);
//                            lane>=16 -> N=lane-16, K = 16+e.
// K index = dt*128 + ci  (dt = time tap, ci = input channel).
// ---------------------------------------------------------------------------
__global__ void prep_w_kernel(const float* __restrict__ coeffs,
                              unsigned short* __restrict__ wfrag) {
  int idx  = blockIdx.x * 256 + threadIdx.x;          // total = 8*36*16*512
  int e    = idx & 15;
  int lane = (idx >> 4) & 31;
  int nt   = (idx >> 9) & 15;
  int rest = idx >> 13;                               // s*NKT + kt
  int kt   = rest % NKT;
  int s    = rest / NKT;

  int k_local = ((lane >> 4) << 4) + e;               // 0..31
  int n  = nt * 16 + (lane & 15);
  int k  = kt * 32 + k_local;
  int dt = k >> 7;                                    // 0..8
  int ci = k & 127;

  float alpha = d_scales[s];
  float p2a   = exp2f(alpha);
  float u     = ((float)dt - 4.0f) * 0.25f / p2a;     // t/half * 2^-alpha
  float gate  = (fabsf(u) <= 1.0f) ? p2a : 0.0f;      // mask * 2^alpha
  float su  = sinf(PI_F * u),        cu  = cosf(PI_F * u);
  float s2u = sinf(2.0f * PI_F * u), c2u = cosf(2.0f * PI_F * u);

  const float* cc = coeffs + (size_t)ci * COUT + n;   // coeffs[k][ci][n], stride CIN*COUT
  const size_t KS = (size_t)CIN * COUT;
  float val = cc[0] + cc[KS]*su + cc[2*KS]*cu + cc[3*KS]*s2u + cc[4*KS]*c2u;
  wfrag[idx] = f2bf(val * gate);
}

// ---------------------------------------------------------------------------
// Fused stage1 (3-tap S conv -> bf16 LDS tile) + stage2 implicit GEMM conv
// via v_wmma_f32_16x16x32_bf16.  Block = 256 thr = 8 waves, 2(M) x 4(N) grid.
// Workgroup output tile: 128 time x 256 cout (f32); wave tile 64 x 64:
// 16 WMMAs per K-iteration against 16 b128 loads.
// ---------------------------------------------------------------------------
__global__ void __launch_bounds__(256)
conv_main_kernel(const float* __restrict__ x,
                 const float* __restrict__ cs,
                 const unsigned short* __restrict__ wfrag,
                 float* __restrict__ out) {
  __shared__ unsigned short xs[(MT + 8) * LDSROW];    // ~37 KB

  const int tid = threadIdx.x;
  const int b   = blockIdx.z;
  const int s   = blockIdx.y;
  const int t0  = blockIdx.x * MT;

  // 3-tap scale kernel from coeffs_scale: basis rows at u = -1, 0, +1
  float c0 = cs[0], c1 = cs[1], c2 = cs[2];
  float w0 = c0 + sinf(-PI_F) * c1 + cosf(-PI_F) * c2;
  float w1 = c0 + c2;
  float w2 = c0 + sinf(PI_F) * c1 + cosf(PI_F) * c2;

  const int sm1 = (s > 0) ? s - 1 : 0;
  const int sp1 = (s < S_ - 1) ? s + 1 : S_ - 1;
  const float* xb = x + (size_t)b * S_ * T_ * CIN;

  // Fill LDS halo tile: rows r -> time t0-4+r, zero outside [0,T) (SAME pad)
  for (int i = tid; i < (MT + 8) * CIN; i += 256) {
    int r = i >> 7, c = i & 127;
    int t = t0 - 4 + r;
    float v = 0.0f;
    if (t >= 0 && t < T_) {
      size_t off = (size_t)t * CIN + c;
      v = w0 * xb[(size_t)sm1 * T_ * CIN + off]
        + w1 * xb[(size_t)s   * T_ * CIN + off]
        + w2 * xb[(size_t)sp1 * T_ * CIN + off];
    }
    xs[r * LDSROW + c] = f2bf(v);
  }
  __syncthreads();

  const int lane = tid & 31;
  const int wid  = tid >> 5;
  const int wm   = wid & 1;        // 2 waves along M (64 rows each)
  const int wn   = wid >> 1;       // 4 waves along N (64 cols each)
  const int half = lane >> 4;
  const int l15  = lane & 15;

  v8f acc[4][4];
#pragma unroll
  for (int mi = 0; mi < 4; ++mi)
#pragma unroll
    for (int ni = 0; ni < 4; ++ni)
      acc[mi][ni] = (v8f){0.f,0.f,0.f,0.f,0.f,0.f,0.f,0.f};

  const unsigned short* wbase = wfrag + (size_t)s * NKT * NNT * 512;

  for (int kt = 0; kt < NKT; ++kt) {
    const int dt  = kt >> 2;
    const int ci0 = (kt & 3) * 32;

    // A fragments from LDS (A 16x32, K order used here (same as B): lane<16 holds K 0..15, lane>=16 holds K 16..31)
    union { v16bf v; uint4 u[2]; } a[4];
#pragma unroll
    for (int mi = 0; mi < 4; ++mi) {
      int r = wm * 64 + mi * 16 + l15 + dt;           // halo row = m + dt
      a[mi].u[0] = *(const uint4*)&xs[r * LDSROW + ci0 + half * 16];
      a[mi].u[1] = *(const uint4*)&xs[r * LDSROW + ci0 + half * 16 + 8];
    }

    // B fragments from pre-swizzled global weights (L2 resident)
    union { v16bf v; uint4 u[2]; } bf[4];
#pragma unroll
    for (int ni = 0; ni < 4; ++ni) {
      int nt = wn * 4 + ni;
      const uint4* p = (const uint4*)(wbase + ((size_t)kt * NNT + nt) * 512 + lane * 16);
      bf[ni].u[0] = p[0];
      bf[ni].u[1] = p[1];
    }

#pragma unroll
    for (int mi = 0; mi < 4; ++mi)
#pragma unroll
      for (int ni = 0; ni < 4; ++ni)
        acc[mi][ni] = __builtin_amdgcn_wmma_f32_16x16x32_bf16(
            false, a[mi].v, false, bf[ni].v, (short)0, acc[mi][ni], false, false);
  }

  // Store: f32 C layout — VGPR r: M = r + 8*(lane>=16), N = lane%16
  float* ob = out + (((size_t)(b * S_ + s)) * T_ + t0) * COUT;
#pragma unroll
  for (int mi = 0; mi < 4; ++mi) {
    int mbase = wm * 64 + mi * 16 + half * 8;
#pragma unroll
    for (int ni = 0; ni < 4; ++ni) {
      int n = wn * 64 + ni * 16 + l15;
#pragma unroll
      for (int r2 = 0; r2 < 8; ++r2)
        ob[(size_t)(mbase + r2) * COUT + n] = acc[mi][ni][r2];
    }
  }
}

extern "C" void kernel_launch(void* const* d_in, const int* in_sizes, int n_in,
                              void* d_out, int out_size, void* d_ws, size_t ws_size,
                              hipStream_t stream) {
  const float* x      = (const float*)d_in[0];   // (8,8,4096,128) f32
  const float* coeffs = (const float*)d_in[1];   // (5,128,256) f32
  const float* cscale = (const float*)d_in[2];   // (3,1,1) f32
  float* out          = (float*)d_out;           // (8,8,4096,256) f32
  unsigned short* wfrag = (unsigned short*)d_ws; // 8*36*16*512 bf16 = ~4.5 MB

  const int wtotal = 8 * NKT * NNT * 512;        // 2,359,296 elems
  prep_w_kernel<<<wtotal / 256, 256, 0, stream>>>(coeffs, wfrag);

  dim3 grid(T_ / MT, S_, 8);                     // 32 x 8 x 8 = 2048 workgroups
  conv_main_kernel<<<grid, 256, 0, stream>>>(x, cscale, wfrag, out);
}
